// UltraOptimizedMoE_36197984371393
// MI455X (gfx1250) — compile-verified, matrix operand order checked by reading, not run on hardware
//
#include <hip/hip_runtime.h>
#include <hip/hip_bf16.h>
#include <math.h>

// ---------------------------------------------------------------------------
// MoE conv block for MI455X (gfx1250, wave32).
// 1x1 shared conv + gated 3x3 expert convs as implicit-GEMM with
// v_wmma_f32_16x16x32_bf16 (f32 accumulate). Fragments are loaded as
// contiguous 16B chunks (ds_load_b128 / global_load_b128) per the ISA 7.12.2
// bf16 operand layout. Each wave owns two M-tiles so every B fragment feeds
// two WMMAs.
// ---------------------------------------------------------------------------

typedef __bf16 bf16;
typedef __attribute__((ext_vector_type(16))) __bf16 v16bf;
typedef __attribute__((ext_vector_type(8)))  __bf16 v8bf;
typedef __attribute__((ext_vector_type(8)))  float  v8f;

#define B_   16
#define C_   256
#define O_   256
#define H_   32
#define W_   32
#define HW_  1024          // 32*32
#define E_   4
#define CR_  16            // C/16
#define G_   8             // groups
#define GCH_ 32            // channels per group
#define IMG_ (C_*HW_)      // 262144 elems per batch image

__device__ __forceinline__ float silu_f(float v) {
    return v / (1.0f + __expf(-v));
}

__device__ __forceinline__ v16bf cat8(v8bf lo, v8bf hi) {
    return __builtin_shufflevector(lo, hi,
        0, 1, 2, 3, 4, 5, 6, 7, 8, 9, 10, 11, 12, 13, 14, 15);
}

// ---------------- f32 -> bf16 conversion (identity layout) ----------------
__global__ void f2bf_kernel(const float* __restrict__ in, bf16* __restrict__ out, int n) {
    int i = blockIdx.x * 256 + threadIdx.x;
    if (i < n) out[i] = (bf16)in[i];
}

// ---------------- expert weights: [E,O,C,3,3] f32 -> [E,9,O,C] bf16 --------
// Makes the GEMM K (=channel) dimension contiguous for b128 B-fragment loads.
__global__ void ewt_kernel(const float* __restrict__ in, bf16* __restrict__ out) {
    int i = blockIdx.x * 256 + threadIdx.x;     // over E*9*O*C
    if (i >= E_ * 9 * O_ * C_) return;
    int c  = i & 255;
    int o  = (i >> 8) & 255;
    int er = i >> 16;                           // e*9 + r
    int r  = er % 9, e = er / 9;
    out[i] = (bf16)in[(((size_t)(e * O_ + o) * C_ + c) * 9) + r];
}

// ---------------- router: pool -> MLP -> softmax -> top-2 gate ----------------
__global__ void router_kernel(const float* __restrict__ x,
                              const float* __restrict__ w1, const float* __restrict__ b1,
                              const float* __restrict__ w2, const float* __restrict__ b2,
                              float* __restrict__ gate) {
    int b = blockIdx.x, t = threadIdx.x;
    __shared__ float pooled[C_];
    __shared__ float hbuf[CR_];
    __shared__ float lg[E_];
    const float* xc = x + (size_t)b * IMG_ + (size_t)t * HW_;
    float s = 0.f;
    for (int i = 0; i < HW_; i += 4) {
        float4 v = *(const float4*)(xc + i);
        s += v.x + v.y + v.z + v.w;
    }
    pooled[t] = s * (1.0f / (float)HW_);
    __syncthreads();
    if (t < CR_) {
        float a = b1[t];
        for (int c = 0; c < C_; ++c) a += pooled[c] * w1[t * C_ + c];
        hbuf[t] = silu_f(a);
    }
    __syncthreads();
    if (t < E_) {
        float a = b2[t];
        for (int j = 0; j < CR_; ++j) a += hbuf[j] * w2[t * CR_ + j];
        lg[t] = a;  // TEMPERATURE == 1
    }
    __syncthreads();
    if (t == 0) {
        float mx = lg[0];
        for (int e = 1; e < E_; ++e) mx = fmaxf(mx, lg[e]);
        float p[E_]; float sum = 0.f;
        for (int e = 0; e < E_; ++e) { p[e] = __expf(lg[e] - mx); sum += p[e]; }
        for (int e = 0; e < E_; ++e) p[e] /= sum;
        int i1 = 0;
        for (int e = 1; e < E_; ++e) if (p[e] > p[i1]) i1 = e;
        int i2 = -1;
        for (int e = 0; e < E_; ++e) { if (e == i1) continue; if (i2 < 0 || p[e] > p[i2]) i2 = e; }
        float ssum = p[i1] + p[i2];
        float wa = p[i1] / ssum, wb = p[i2] / ssum;
        if (wa < 0.01f) wa = 0.f;
        if (wb < 0.01f) wb = 0.f;
        float g[E_] = {0.f, 0.f, 0.f, 0.f};
        g[i1] = wa; g[i2] = wb;
        for (int e = 0; e < E_; ++e) gate[b * E_ + e] = g[e];
    }
}

// ---------------- implicit-GEMM conv via WMMA bf16 ----------------
// Block = 256 threads = 8 waves. Block tile: M=32 (one full image row),
// N=128 out-channels; wave wv owns N-strip wv*16 with TWO 16x16 M-tiles
// (two f32 accumulators). Outer loop: 8 channel-groups of 32; x slab
// (32ch x ROWS x COLS, channel innermost) staged in LDS once per group.
// Inner unrolled loop over TAPS: B = 2x global_load_b128 (weights [E,9,O,C]),
// A = 2x2 ds_load_b128, 2x v_wmma_f32_16x16x32_bf16 (B reused).
template <int TAPS>
__global__ void conv_wmma_kernel(const bf16* __restrict__ xb,
                                 const bf16* __restrict__ wt,
                                 const float* __restrict__ gate, int e,
                                 float* __restrict__ y) {
    const int b = blockIdx.z;
    if (TAPS == 9) {
        if (gate[b * E_ + e] <= 0.f) return;   // uniform per block: safe early exit
    }
    constexpr int ROFF = (TAPS == 9) ? 1 : 0;       // halo offset
    constexpr int ROWS = (TAPS == 9) ? 3 : 1;
    constexpr int COLS = 32 + 2 * ROFF;             // 34 or 32
    constexpr int NSTG = ROWS * COLS * 32;          // staged elements

    const int lane = threadIdx.x & 31;
    const int wv   = threadIdx.x >> 5;
    const int hrow = blockIdx.x;                    // image row, M block = 32
    const int n0   = blockIdx.y * 128 + wv * 16;    // this wave's N strip

    const int mloc = lane & 15;
    const int hi8  = (lane >> 4) << 3;              // ISA A/B K-layout offset
    const int nch  = n0 + mloc;

    __shared__ __align__(16) bf16 a_slab[ROWS * 34 * 32];   // K innermost

    v8f acc0 = {};                                  // M-tile 0 (cols 0..15)
    v8f acc1 = {};                                  // M-tile 1 (cols 16..31)
    const bf16* xbase = xb + (size_t)b * IMG_;

    for (int cg = 0; cg < 8; ++cg) {
        const int c0 = cg << 5;

        // ---- cooperative slab staging (col innermost -> coalesced reads) ----
        __syncthreads();
        for (int s = threadIdx.x; s < NSTG; s += 256) {
            int col = s % COLS;
            int rc  = s / COLS;                      // row*32 + ci
            int ci  = rc & 31;
            int row = rc >> 5;
            int hh  = hrow + row - ROFF;
            int ww  = col - ROFF;
            bf16 v = (bf16)0.0f;
            if ((unsigned)hh < (unsigned)H_ && (unsigned)ww < (unsigned)W_)
                v = xbase[((c0 + ci) << 10) + (hh << 5) + ww];
            a_slab[(row * 34 + col) * 32 + ci] = v;
        }
        __syncthreads();

        // prefetch next channel-group's weight run (global_prefetch_b8)
        if (cg < 7) {
            const bf16* wpn = wt + (((size_t)(e * TAPS) * O_ + nch) * C_ + c0 + 32);
            __builtin_prefetch(wpn, 0, 0);
        }

        // ---- 9 (or 1) taps, fully unrolled; B feeds both M-tiles ----
        #pragma unroll
        for (int r = 0; r < TAPS; ++r) {
            const int dh = (TAPS == 9) ? (r / 3 - 1) : 0;
            const int dw = (TAPS == 9) ? (r % 3 - 1) : 0;

            // B fragment: contiguous channels from [E,TAPS,O,C] bf16 weights
            const bf16* wp = wt + (((size_t)(e * TAPS + r) * O_ + nch) * C_ + c0);
            v16bf bm = cat8(*(const v8bf*)(wp + hi8),
                            *(const v8bf*)(wp + 16 + hi8));

            // A fragments for the two M-tiles (per-lane two 16B runs each)
            const bf16* arow = a_slab + ((size_t)((dh + ROFF) * 34 + (dw + ROFF))) * 32;
            const bf16* ap0  = arow + (size_t)mloc * 32;
            const bf16* ap1  = arow + (size_t)(mloc + 16) * 32;
            v16bf a0 = cat8(*(const v8bf*)(ap0 + hi8),
                            *(const v8bf*)(ap0 + 16 + hi8));
            v16bf a1 = cat8(*(const v8bf*)(ap1 + hi8),
                            *(const v8bf*)(ap1 + 16 + hi8));

            acc0 = __builtin_amdgcn_wmma_f32_16x16x32_bf16(
                false, a0, false, bm, (short)0, acc0, false, false);
            acc1 = __builtin_amdgcn_wmma_f32_16x16x32_bf16(
                false, a1, false, bm, (short)0, acc1, false, false);
        }
    }

    // D layout: VGPR r8, lanes 0-15 -> M=r8, N=lane; lanes 16-31 -> M=r8+8.
    // Consecutive r8 -> consecutive spatial -> 16B stores.
    float* yb = y + ((size_t)b * O_ + nch) * HW_ + (hrow << 5) + hi8;
    *(float4*)(yb)          = make_float4(acc0[0], acc0[1], acc0[2], acc0[3]);
    *(float4*)(yb + 4)      = make_float4(acc0[4], acc0[5], acc0[6], acc0[7]);
    *(float4*)(yb + 16)     = make_float4(acc1[0], acc1[1], acc1[2], acc1[3]);
    *(float4*)(yb + 20)     = make_float4(acc1[4], acc1[5], acc1[6], acc1[7]);
}

// ---------------- GroupNorm stats (mu, rsigma) per (batch, group) ----------------
__global__ void gn_stats_kernel(const float* __restrict__ y,
                                const float* __restrict__ gate, int e,
                                float* __restrict__ stats) {
    int g = blockIdx.x & 7, b = blockIdx.x >> 3;
    if (e >= 0 && gate[b * E_ + e] <= 0.f) return;
    const float* base = y + (size_t)b * IMG_ + (size_t)g * (GCH_ * HW_);
    float s = 0.f, s2 = 0.f;
    for (int i = threadIdx.x * 4; i < GCH_ * HW_; i += 1024) {
        float4 v = *(const float4*)(base + i);
        s  += v.x + v.y + v.z + v.w;
        s2 += v.x * v.x + v.y * v.y + v.z * v.z + v.w * v.w;
    }
    __shared__ float rs[256], rq[256];
    rs[threadIdx.x] = s; rq[threadIdx.x] = s2;
    __syncthreads();
    for (int off = 128; off > 0; off >>= 1) {
        if (threadIdx.x < off) {
            rs[threadIdx.x] += rs[threadIdx.x + off];
            rq[threadIdx.x] += rq[threadIdx.x + off];
        }
        __syncthreads();
    }
    if (threadIdx.x == 0) {
        const float inv = 1.0f / (float)(GCH_ * HW_);
        float mu  = rs[0] * inv;
        float var = rq[0] * inv - mu * mu;
        stats[(b * G_ + g) * 2 + 0] = mu;
        stats[(b * G_ + g) * 2 + 1] = rsqrtf(var + 1e-5f);
    }
}

// ---------------- GN-affine + SiLU epilogue; write (shared) or gated add ----
__global__ void combine_kernel(const float* __restrict__ y,
                               const float* __restrict__ stats,
                               const float* __restrict__ gamma,
                               const float* __restrict__ beta,
                               const float* __restrict__ gate, int e,
                               float* __restrict__ out) {
    size_t i = ((size_t)blockIdx.x * 256 + threadIdx.x) * 4;   // over B*O*H*W
    int b = (int)(i >> 18);
    int o = (int)((i >> 10) & 255);
    int g = o >> 5;
    float gt = 1.f;
    if (e >= 0) { gt = gate[b * E_ + e]; if (gt <= 0.f) return; }
    float mu = stats[(b * G_ + g) * 2 + 0];
    float rg = stats[(b * G_ + g) * 2 + 1];
    float gm = gamma[o], bt = beta[o];
    float4 v = *(const float4*)(y + i);
    float4 r;
    r.x = silu_f((v.x - mu) * rg * gm + bt);
    r.y = silu_f((v.y - mu) * rg * gm + bt);
    r.z = silu_f((v.z - mu) * rg * gm + bt);
    r.w = silu_f((v.w - mu) * rg * gm + bt);
    if (e >= 0) {
        float4 o4 = *(const float4*)(out + i);
        o4.x += gt * r.x; o4.y += gt * r.y; o4.z += gt * r.z; o4.w += gt * r.w;
        *(float4*)(out + i) = o4;
    } else {
        *(float4*)(out + i) = r;
    }
}

// ---------------------------------------------------------------------------
extern "C" void kernel_launch(void* const* d_in, const int* in_sizes, int n_in,
                              void* d_out, int out_size, void* d_ws, size_t ws_size,
                              hipStream_t stream) {
    const float* x        = (const float*)d_in[0];
    const float* rw1      = (const float*)d_in[1];
    const float* rb1      = (const float*)d_in[2];
    const float* rw2      = (const float*)d_in[3];
    const float* rb2      = (const float*)d_in[4];
    const float* ew       = (const float*)d_in[5];
    const float* eg_gamma = (const float*)d_in[6];
    const float* eg_beta  = (const float*)d_in[7];
    const float* sw       = (const float*)d_in[8];
    const float* sg_gamma = (const float*)d_in[9];
    const float* sg_beta  = (const float*)d_in[10];
    float* out = (float*)d_out;

    const int NX  = B_ * C_ * HW_;        // 4,194,304
    const int NEW = E_ * O_ * C_ * 9;     // 2,359,296
    const int NSW = O_ * C_;              // 65,536

    char* ws = (char*)d_ws;
    bf16*  xb    = (bf16*)(ws);                         // 8,388,608 B
    bf16*  ewb   = (bf16*)(ws + 8388608);               // 4,718,592 B  [E,9,O,C]
    bf16*  swb   = (bf16*)(ws + 13107200);              // 131,072 B    [O,C]
    float* gate  = (float*)(ws + 13238272);             // 256 B
    float* stats = (float*)(ws + 13238528);             // 1,024 B
    float* conv  = (float*)(ws + 13240320);             // 16,777,216 B

    // precision/layout conversion for WMMA operands
    f2bf_kernel<<<(NX  + 255) / 256, 256, 0, stream>>>(x,  xb,  NX);
    ewt_kernel <<<(NEW + 255) / 256, 256, 0, stream>>>(ew, ewb);
    f2bf_kernel<<<(NSW + 255) / 256, 256, 0, stream>>>(sw, swb, NSW);

    // router -> gate[B,E]
    router_kernel<<<B_, 256, 0, stream>>>(x, rw1, rb1, rw2, rb2, gate);

    dim3 cgrid(H_, O_ / 128, B_);   // (32, 2, 16), 256 threads = 8 waves

    // shared expert: 1x1 conv -> GN stats -> out = silu(gn(..))
    conv_wmma_kernel<1><<<cgrid, 256, 0, stream>>>(xb, swb, gate, 0, conv);
    gn_stats_kernel<<<B_ * G_, 256, 0, stream>>>(conv, gate, -1, stats);
    combine_kernel<<<NX / 1024, 256, 0, stream>>>(conv, stats, sg_gamma, sg_beta,
                                                  gate, -1, out);

    // sparse experts: gated 3x3 conv -> GN stats -> out += gate * silu(gn(..))
    for (int e = 0; e < E_; ++e) {
        conv_wmma_kernel<9><<<cgrid, 256, 0, stream>>>(xb, ewb, gate, e, conv);
        gn_stats_kernel<<<B_ * G_, 256, 0, stream>>>(conv, gate, e, stats);
        combine_kernel<<<NX / 1024, 256, 0, stream>>>(conv, stats,
                                                      eg_gamma + e * O_,
                                                      eg_beta  + e * O_,
                                                      gate, e, out);
    }
}
